// DecoderFG_10041633538400
// MI455X (gfx1250) — matrix operand order repaired and checked by reading, not compiled, not run on hardware
//
#include <hip/hip_runtime.h>
#include <hip/hip_bf16.h>
#include <math.h>

typedef _Float16 h16;
typedef __attribute__((ext_vector_type(8)))  _Float16 v8h;
typedef __attribute__((ext_vector_type(16))) _Float16 v16h;
typedef __attribute__((ext_vector_type(8)))  float    v8f;

#define K_SLOTS   8
#define P_PTS     131072
#define NROWS     (K_SLOTS * P_PTS)
#define TILE_ROWS 256                     // 8 waves x 32 rows
#define NTILES    (NROWS / TILE_ROWS)     // 4096
#define LOC       0.5714285714285714f     // 4/7

// ---- d_out layout (float offsets), tuple return order:
// raws [P,4], masked_raws [K,P,4], unmasked_raws [K,P,4], masks [K,P]
#define OFF_RAWS   0
#define OFF_MASKED (P_PTS * 4)
#define OFF_UNM    (P_PTS * 4 + K_SLOTS * P_PTS * 4)
#define OFF_MASKS  (P_PTS * 4 + 2 * K_SLOTS * P_PTS * 4)

// ---- LDS layout ----
// f16 weight region (offsets in halves); K-dims zero-padded to mult of 32.
// Concat layer uses remapped input columns: [x 0..96 | pad | h 112..175 | pad].
#define HW_B0   0        // 64 x 128 (97 real)
#define HW_B1   8192     // 64 x 64
#define HW_B2   12288    // 64 x 64
#define HW_A0   16384    // 64 x 192 (97 + 64 real, remapped)
#define HW_A1   28672    // 64 x 64
#define HW_A2   32768    // 64 x 64
#define HW_LAT  36864    // 64 x 64
#define HW_C0   40960    // 16 x 64
#define HW_TOT  41984
// f32 region (offsets in floats), base = HW_TOT*2 bytes
#define FB_B0   0
#define FB_B1   64
#define FB_B2   128
#define FB_A0   192
#define FB_A1   256
#define FB_A2   320
#define FB_LAT  384
#define FB_C0   448      // 16
#define FW_C1   464      // 48 (= w_c1 [3][16])
#define FB_C1   512      // 3
#define FW_SH   515      // 64 (= w_shape row)
#define FB_SH   579      // 1
#define FT      580      // 9  (= 3x3 transform)
#define F_TOT   592
#define ACT_BASE_BYTES (HW_TOT * 2 + F_TOT * 4)   // 86336, 32B aligned
// per-wave activation block (bytes):
// act0 32x192 h | act1 32x64 h | act2 32x64 h | cbuf 32x16 f32
#define ACT0_OFF 0
#define ACT1_OFF 12288
#define ACT2_OFF 16384
#define CBUF_OFF 20480
#define WAVE_ACT_BYTES 22528
#define LDS_BYTES (ACT_BASE_BYTES + 8 * WAVE_ACT_BYTES)  // 266560 (< 320KB WGP)

#define COL_H   112      // concat h-block column base (16B aligned for b128 stores)

// ---------------- WMMA fragment helpers (CDNA5 16x16x32 f16 layouts) ----------------
__device__ __forceinline__ v16h make16(v8h a, v8h b) {
    union { v8h h[2]; v16h v; } u; u.h[0] = a; u.h[1] = b; return u.v;
}

// A-matrix = WEIGHTS (16 outs x 32 K), row-major [out][in] in LDS, row stride ldw.
// lanes 0-15: M=lane, K=k0+0..7 / k0+16..23 ; lanes 16-31: M=lane-16, K=k0+8..15 / k0+24..31
__device__ __forceinline__ v16h ldsA_w(const h16* w, int ldw, int lane, int mt, int k0) {
    const int m  = (mt << 4) + (lane & 15);
    const int hi = (lane >> 4) & 1;
    const h16* rp = w + m * ldw + k0;
    v8h a = *(const v8h*)(rp + (hi ? 8 : 0));
    v8h b = *(const v8h*)(rp + (hi ? 24 : 16));
    return make16(a, b);
}

// B-matrix = ACTIVATIONS (32 K x 16 rows): column n = activation row n, contiguous in K.
// lanes 0-15: col=lane, K=k0+0..7 / k0+16..23 ; lanes 16-31: col=lane-16, K=k0+8..15 / k0+24..31
__device__ __forceinline__ v16h ldsB_x(const h16* act, int lda, int lane, int k0) {
    const int r  = lane & 15;
    const int hi = (lane >> 4) & 1;
    const h16* rp = act + r * lda + k0 + (hi ? 8 : 0);
    v8h a = *(const v8h*)rp;
    v8h b = *(const v8h*)(rp + 16);
    return make16(a, b);
}

// D layout now M=out, N=row: lane holds 8 CONTIGUOUS output features of one row
// -> single packed 16B store, layout matches next layer's B read directly.
__device__ __forceinline__ void storeD8(h16* oBuf, int ldo, int colBase, int lane,
                                        v8f acc, bool doRelu) {
    const int row = lane & 15;
    v8h hv;
#pragma unroll
    for (int r = 0; r < 8; ++r) {
        float v = acc[r];
        if (doRelu) v = fmaxf(v, 0.f);
        hv[r] = (h16)v;
    }
    *(v8h*)(oBuf + row * ldo + colBase) = hv;
}

// One 32-row layer, D = W * X^T. Bias rides the WMMA C operand (v8f from LDS).
// Each weight A-fragment feeds 2 WMMAs (two row groups); activations B reused across MT.
template <int KCH, int MT>
__device__ __forceinline__ void mlp_layer(const h16* aBuf, int lda,
                                          const h16* wBuf, int ldw,
                                          const float* bias,
                                          h16* oBuf, int ldo, int colOff,
                                          int lane, bool doRelu) {
    v16h B0[KCH], B1[KCH];
#pragma unroll
    for (int kc = 0; kc < KCH; ++kc) {
        B0[kc] = ldsB_x(aBuf,            lda, lane, kc * 32);
        B1[kc] = ldsB_x(aBuf + 16 * lda, lda, lane, kc * 32);
    }
    const int hi8 = ((lane >> 4) & 1) * 8;
#pragma unroll
    for (int mt = 0; mt < MT; ++mt) {
        const v8f bias8 = *(const v8f*)(bias + (mt << 4) + hi8);  // C = bias (free add)
        v8f acc0 = bias8, acc1 = bias8;
#pragma unroll
        for (int kc = 0; kc < KCH; ++kc) {
            v16h Aw = ldsA_w(wBuf, ldw, lane, mt, kc * 32);
            acc0 = __builtin_amdgcn_wmma_f32_16x16x32_f16(
                false, Aw, false, B0[kc], (short)0, acc0, false, false);
            acc1 = __builtin_amdgcn_wmma_f32_16x16x32_f16(
                false, Aw, false, B1[kc], (short)0, acc1, false, false);
        }
        const int colBase = colOff + (mt << 4) + hi8;
        storeD8(oBuf,            ldo, colBase, lane, acc0, doRelu);
        storeD8(oBuf + 16 * ldo, ldo, colBase, lane, acc1, doRelu);
    }
}

// c0 head: 64 -> 16, ReLU, both row groups, f32 result [32][16] (packed 32B stores).
__device__ __forceinline__ void layer_c0(const h16* aBuf, const h16* wc0,
                                         const float* bc0, float* cbuf, int lane) {
    const int hi8 = ((lane >> 4) & 1) * 8;
    const v8f bias8 = *(const v8f*)(bc0 + hi8);
    v8f acc0 = bias8, acc1 = bias8;
#pragma unroll
    for (int kc = 0; kc < 2; ++kc) {
        v16h Aw = ldsA_w(wc0, 64, lane, 0, kc * 32);
        v16h b0 = ldsB_x(aBuf,           64, lane, kc * 32);
        v16h b1 = ldsB_x(aBuf + 16 * 64, 64, lane, kc * 32);
        acc0 = __builtin_amdgcn_wmma_f32_16x16x32_f16(false, Aw, false, b0, (short)0, acc0, false, false);
        acc1 = __builtin_amdgcn_wmma_f32_16x16x32_f16(false, Aw, false, b1, (short)0, acc1, false, false);
    }
    const int row = lane & 15;
    v8f r0, r1;
#pragma unroll
    for (int r = 0; r < 8; ++r) { r0[r] = fmaxf(acc0[r], 0.f); r1[r] = fmaxf(acc1[r], 0.f); }
    *(v8f*)(cbuf + row * 16 + hi8)        = r0;
    *(v8f*)(cbuf + (row + 16) * 16 + hi8) = r1;
}

// f32 global weight -> f16 LDS with zero K-padding
__device__ __forceinline__ void loadW(h16* dst, const float* src, int outs,
                                      int realK, int padK, int tid, int nthr) {
    const int tot = outs * padK;
    for (int i = tid; i < tot; i += nthr) {
        const int o  = i / padK;
        const int kk = i - o * padK;
        dst[i] = (h16)((kk < realK) ? src[o * realK + kk] : 0.f);
    }
}

// w_a0 [64][161] staged with remapped columns: 0..96 -> 0..96, 97..160 -> 112..175, else 0
__device__ __forceinline__ void loadWa0(h16* dst, const float* src, int tid, int nthr) {
    for (int i = tid; i < 64 * 192; i += nthr) {
        const int o = i / 192;
        const int c = i - o * 192;
        float v = 0.f;
        if (c < 97)                      v = src[o * 161 + c];
        else if (c >= COL_H && c < COL_H + 64) v = src[o * 161 + (c - (COL_H - 97))];
        dst[i] = (h16)v;
    }
}

// ---------------- Pass 1: per-row MLP via WMMA, writes unmasked_raws ----------------
__global__ __launch_bounds__(256) void decoder_pass1(
    const float* __restrict__ sc,   const float* __restrict__ zslots,
    const float* __restrict__ ftr,  const float* __restrict__ spos,
    const float* wb0, const float* bb0, const float* wb1, const float* bb1,
    const float* wb2, const float* bb2, const float* wa0, const float* ba0,
    const float* wa1, const float* ba1, const float* wa2, const float* ba2,
    const float* wlat, const float* blat, const float* wsh, const float* bsh,
    const float* wc0, const float* bc0, const float* wc1, const float* bc1,
    float* __restrict__ out) {
    extern __shared__ char smem[];
    h16*   hw = (h16*)smem;
    float* fw = (float*)(smem + HW_TOT * 2);
    char*  actbase = smem + ACT_BASE_BYTES;

    const int tid  = threadIdx.x;
    const int lane = tid & 31;
    const int wid  = tid >> 5;
    const int nthr = blockDim.x;

    // ---- stage weights into LDS (f32 -> f16, zero-padded) ----
    loadW(hw + HW_B0,  wb0,  64, 97,  128, tid, nthr);
    loadW(hw + HW_B1,  wb1,  64, 64,  64,  tid, nthr);
    loadW(hw + HW_B2,  wb2,  64, 64,  64,  tid, nthr);
    loadWa0(hw + HW_A0, wa0, tid, nthr);
    loadW(hw + HW_A1,  wa1,  64, 64,  64,  tid, nthr);
    loadW(hw + HW_A2,  wa2,  64, 64,  64,  tid, nthr);
    loadW(hw + HW_LAT, wlat, 64, 64,  64,  tid, nthr);
    loadW(hw + HW_C0,  wc0,  16, 64,  64,  tid, nthr);
    for (int i = tid; i < 64; i += nthr) {
        fw[FB_B0 + i] = bb0[i];  fw[FB_B1 + i] = bb1[i];  fw[FB_B2 + i] = bb2[i];
        fw[FB_A0 + i] = ba0[i];  fw[FB_A1 + i] = ba1[i];  fw[FB_A2 + i] = ba2[i];
        fw[FB_LAT + i] = blat[i]; fw[FW_SH + i] = wsh[i];
    }
    for (int i = tid; i < 48; i += nthr) fw[FW_C1 + i] = wc1[i];
    if (tid < 16) fw[FB_C0 + tid] = bc0[tid];
    if (tid < 9)  fw[FT + tid]    = ftr[tid];
    if (tid < 3)  fw[FB_C1 + tid] = bc1[tid];
    if (tid == 0) fw[FB_SH]       = bsh[0];
    __syncthreads();

    h16*   act0 = (h16*)(actbase + wid * WAVE_ACT_BYTES + ACT0_OFF); // [32][192]
    h16*   act1 = (h16*)(actbase + wid * WAVE_ACT_BYTES + ACT1_OFF); // [32][64]
    h16*   act2 = (h16*)(actbase + wid * WAVE_ACT_BYTES + ACT2_OFF); // [32][64]
    float* cbuf = (float*)(actbase + wid * WAVE_ACT_BYTES + CBUF_OFF); // [32][16]

    // zero act0 pad columns 97..191 ONCE (only 0..96 and 112..175 are ever rewritten;
    // keeps padded WMMA K-chunks exact and avoids NaN*0 from uninitialized LDS)
    for (int i = lane; i < 32 * 95; i += 32)
        act0[(i / 95) * 192 + 97 + (i % 95)] = (h16)0.f;

    const float T00 = fw[FT + 0], T01 = fw[FT + 1], T02 = fw[FT + 2];
    const float T10 = fw[FT + 3], T11 = fw[FT + 4], T12 = fw[FT + 5];
    const float T20 = fw[FT + 6], T21 = fw[FT + 7], T22 = fw[FT + 8];

    for (int tile = blockIdx.x; tile < NTILES; tile += gridDim.x) {
        const int rowG0 = tile * TILE_ROWS + wid * 32;  // wave's 32 global rows
        const int kSlot = rowG0 >> 17;                  // P = 2^17; tile never spans slots

        // ---- build input_fg row for this lane (all 32 lanes: one row each) ----
        bool outsider;
        {
            const float* cp = sc + (size_t)(rowG0 + lane) * 3;
            const float x = cp[0], y = cp[1], z = cp[2];
            const float tx = T00 * x + T01 * y + T02 * z;
            const float ty = T10 * x + T11 * y + T12 * z;
            const float tz = T20 * x + T21 * y + T22 * z;
            outsider = (fabsf(tx) > LOC) || (fabsf(ty) > LOC) || (fabsf(tz) > LOC);
            const float sx = spos[kSlot * 3 + 0], sy = spos[kSlot * 3 + 1], sz = spos[kSlot * 3 + 2];
            const float rx = x - sx, ry = y - sy, rz = z - sz;
            const float cx = T00 * rx + T01 * ry + T02 * rz;
            const float cy = T10 * rx + T11 * ry + T12 * rz;
            const float cz = T20 * rx + T21 * ry + T22 * rz;
            h16* rp = act0 + lane * 192;
            rp[0] = (h16)cx; rp[1] = (h16)cy; rp[2] = (h16)cz;
            float f = 1.f;
#pragma unroll
            for (int j = 0; j < 5; ++j) {  // [x, sin(fx), cos(fx)] blocks, f = 1,2,4,8,16
                rp[3 + j * 6 + 0] = (h16)__sinf(f * cx);
                rp[3 + j * 6 + 1] = (h16)__sinf(f * cy);
                rp[3 + j * 6 + 2] = (h16)__sinf(f * cz);
                rp[6 + j * 6 + 0] = (h16)__cosf(f * cx);
                rp[6 + j * 6 + 1] = (h16)__cosf(f * cy);
                rp[6 + j * 6 + 2] = (h16)__cosf(f * cz);
                f *= 2.f;
            }
        }
        // z-slot features (cols 33..96)
        const float* zp = zslots + kSlot * 64;
        for (int i = lane; i < 32 * 64; i += 32)
            act0[(i >> 6) * 192 + 33 + (i & 63)] = (h16)zp[i & 63];

        // ---- MLP chain, D = W * X^T (164 WMMAs / wave / tile) ----
        mlp_layer<4, 4>(act0, 192, hw + HW_B0, 128, fw + FB_B0, act1, 64, 0,  lane, true);
        mlp_layer<2, 4>(act1, 64,  hw + HW_B1, 64,  fw + FB_B1, act2, 64, 0,  lane, true);
        mlp_layer<2, 4>(act2, 64,  hw + HW_B2, 64,  fw + FB_B2, act0, 192, COL_H, lane, true); // concat
        mlp_layer<6, 4>(act0, 192, hw + HW_A0, 192, fw + FB_A0, act1, 64, 0,  lane, true);
        mlp_layer<2, 4>(act1, 64,  hw + HW_A1, 64,  fw + FB_A1, act2, 64, 0,  lane, true);
        mlp_layer<2, 4>(act2, 64,  hw + HW_A2, 64,  fw + FB_A2, act1, 64, 0,  lane, true);  // tmp
        mlp_layer<2, 4>(act1, 64,  hw + HW_LAT, 64, fw + FB_LAT, act2, 64, 0, lane, false); // latent
        layer_c0(act2, hw + HW_C0, fw + FB_C0, cbuf, lane);

        // ---- tiny heads + write unmasked_raws (one row per lane) ----
        {
            const h16* tr = act1 + lane * 64;  // tmp row
            float sh = fw[FB_SH];
#pragma unroll
            for (int i = 0; i < 64; ++i) sh += (float)tr[i] * fw[FW_SH + i];
            if (outsider) sh = 0.f;
            const float sigma = fmaxf(sh, 0.f);
            const float* cb = cbuf + lane * 16;
            float rgb[3];
#pragma unroll
            for (int j = 0; j < 3; ++j) {
                float a = fw[FB_C1 + j];
#pragma unroll
                for (int i = 0; i < 16; ++i) a += cb[i] * fw[FW_C1 + j * 16 + i];
                rgb[j] = (tanhf(a) + 1.f) * 0.5f;
            }
            float4 o; o.x = rgb[0]; o.y = rgb[1]; o.z = rgb[2]; o.w = sigma;
            *(float4*)(out + OFF_UNM + (size_t)(rowG0 + lane) * 4) = o;
        }
    }
}

// ---------------- Pass 2: cross-slot mask normalization + outputs ----------------
__global__ __launch_bounds__(256) void decoder_pass2(float* __restrict__ out) {
    const int p = blockIdx.x * 256 + threadIdx.x;
    if (p >= P_PTS) return;
    float4 u[K_SLOTS];
    float s = 0.f;
#pragma unroll
    for (int k = 0; k < K_SLOTS; ++k) {
        u[k] = *(const float4*)(out + OFF_UNM + ((size_t)k * P_PTS + p) * 4);
        s += u[k].w;  // raw density per slot
    }
    const float inv = 1.f / (s + 1e-5f);
    float4 acc; acc.x = acc.y = acc.z = acc.w = 0.f;
#pragma unroll
    for (int k = 0; k < K_SLOTS; ++k) {
        const size_t idx = (size_t)k * P_PTS + p;
        const float mk = u[k].w * inv;
        out[OFF_MASKS + idx] = mk;
        float4 md;
        md.x = u[k].x * mk; md.y = u[k].y * mk; md.z = u[k].z * mk; md.w = u[k].w * mk;
        *(float4*)(out + OFF_MASKED + idx * 4) = md;
        acc.x += md.x; acc.y += md.y; acc.z += md.z; acc.w += md.w;
    }
    *(float4*)(out + OFF_RAWS + (size_t)p * 4) = acc;
}

extern "C" void kernel_launch(void* const* d_in, const int* in_sizes, int n_in,
                              void* d_out, int out_size, void* d_ws, size_t ws_size,
                              hipStream_t stream) {
    (void)in_sizes; (void)n_in; (void)out_size; (void)d_ws; (void)ws_size;
    const float* A[24];
    for (int i = 0; i < 24; ++i) A[i] = (const float*)d_in[i];
    float* out = (float*)d_out;

    decoder_pass1<<<1024, 256, LDS_BYTES, stream>>>(
        A[0], A[1], A[2], A[3], A[4], A[5], A[6], A[7], A[8], A[9], A[10], A[11],
        A[12], A[13], A[14], A[15], A[16], A[17], A[18], A[19], A[20], A[21], A[22], A[23],
        out);
    decoder_pass2<<<(P_PTS + 255) / 256, 256, 0, stream>>>(out);
}
